// WeightOnlyBlockwiseQuantizedLinear_76639396430015
// MI455X (gfx1250) — matrix operand order repaired and checked by reading, not compiled, not run on hardware
//
#include <hip/hip_runtime.h>
#include <stdint.h>

typedef __attribute__((ext_vector_type(16))) _Float16    v16h;
typedef __attribute__((ext_vector_type(8)))  _Float16    v8h;
typedef __attribute__((ext_vector_type(4)))  _Float16    v4h;
typedef __attribute__((ext_vector_type(8)))  float       v8f;
typedef __attribute__((ext_vector_type(16))) signed char vc16;

// Problem dims (fixed by the reference)
constexpr int BATCH = 4, SEQ = 2048, IN_F = 4096, OUT_F = 4096;
constexpr int M_TOTAL = BATCH * SEQ;   // 8192
constexpr int K = IN_F;                // 4096
constexpr int N = OUT_F;               // 4096
constexpr int QBLOCK = 128;            // quantization block along K

// Tile config
constexpr int BM = 128;                // M per workgroup
constexpr int BN = 128;                // N per workgroup
constexpr int BK = 64;                 // K staged per iteration (half a quant block)
constexpr int APITCH = 72;             // LDS row pitch in halfs (144B: 16B-aligned, bank-stride 36 dwords)
constexpr int WPITCH = 72;

__global__ __launch_bounds__(256)
void wq_gemm_wmma(const float* __restrict__ x,
                  const int8_t* __restrict__ w,        // (K, N) int8 (nb*128 flattened)
                  const float* __restrict__ scaler,    // (K/128, N)
                  float* __restrict__ out)
{
    __shared__ _Float16 As[BM * APITCH];   // [m][k] f16, 18432 B
    __shared__ _Float16 Wt[BN * WPITCH];   // [n][k] f16 (transposed, dequantized), 18432 B

    const int tid   = threadIdx.x;
    const int lane  = tid & 31;
    const int wid   = tid >> 5;          // 8 waves
    const int wave_m = wid & 3;          // 4 waves along M (32 rows each)
    const int wave_n = wid >> 2;         // 2 waves along N (64 cols each)

    const int n0 = blockIdx.x * BN;
    const int m0 = blockIdx.y * BM;

    v8f acc[2][4] = {};                  // 2 M-frags x 4 N-frags of 16x16 f32

    // W staging mapping: each thread owns 2 adjacent k rows x 16 n columns
    const int r2  = tid >> 3;            // 0..31 -> k pair (2*r2, 2*r2+1)
    const int seg = tid & 7;             // 16-column segment within BN

    float sc[16];

    for (int blk = 0; blk < K / QBLOCK; ++blk) {
        // Per-block, per-n scales for this thread's fixed 16-column segment
        const float4* sp = (const float4*)(scaler + (size_t)blk * N + n0 + seg * 16);
        #pragma unroll
        for (int j = 0; j < 4; ++j) {
            float4 f = sp[j];
            sc[4 * j + 0] = f.x; sc[4 * j + 1] = f.y;
            sc[4 * j + 2] = f.z; sc[4 * j + 3] = f.w;
        }

        #pragma unroll 1
        for (int cc = 0; cc < 2; ++cc) {
            const int k0 = blk * QBLOCK + cc * BK;

            __syncthreads();   // previous compute done before LDS overwrite

            // ---- Stage A: 128x64 fp32 -> f16 in LDS ----
            #pragma unroll
            for (int i = 0; i < 8; ++i) {
                const int idx  = tid + i * 256;
                const int row  = idx >> 4;        // 0..127
                const int quad = idx & 15;        // float4 within 64 K
                float4 f = *(const float4*)(x + (size_t)(m0 + row) * K + k0 + quad * 4);
                v4h h;
                h[0] = (_Float16)f.x; h[1] = (_Float16)f.y;
                h[2] = (_Float16)f.z; h[3] = (_Float16)f.w;
                *(v4h*)&As[row * APITCH + quad * 4] = h;
            }

            // ---- Stage W: 64x128 int8 -> dequant f16, transposed [n][k] ----
            {
                const int kr0 = 2 * r2;                       // even k row in chunk
                const int8_t* wp = w + (size_t)(k0 + kr0) * N + n0 + seg * 16;
                vc16 q0 = *(const vc16*)(wp);
                vc16 q1 = *(const vc16*)(wp + N);
                #pragma unroll
                for (int j = 0; j < 16; ++j) {
                    const int nl = seg * 16 + j;
                    _Float16* d = &Wt[nl * WPITCH + kr0];
                    d[0] = (_Float16)(sc[j] * (float)q0[j]);  // adjacent halves ->
                    d[1] = (_Float16)(sc[j] * (float)q1[j]);  // merged b32 store
                }
            }

            // Pull next K-chunk toward caches while we compute
            if (k0 + BK < K) {
                __builtin_prefetch(x + (size_t)(m0 + (tid >> 1)) * K + k0 + BK + (tid & 1) * 32, 0, 1);
                __builtin_prefetch(w + (size_t)(k0 + BK + (tid >> 2)) * N + n0 + (tid & 3) * 32, 0, 1);
            }

            __syncthreads();   // staging visible to all waves

            // ---- Compute: two 16x16x32 K-steps over this 64-chunk ----
            const int khalf = lane >> 4;
            const int lrow  = lane & 15;
            #pragma unroll
            for (int kk = 0; kk < BK; kk += 32) {
                // A fragments: 16x32 f16 per ISA layout
                // lanes 0-15: V0-3 = K0..7,  V4-7 = K16..23 ; lanes 16-31: +8
                v16h aF[2];
                #pragma unroll
                for (int im = 0; im < 2; ++im) {
                    const _Float16* pa =
                        &As[(wave_m * 32 + im * 16 + lrow) * APITCH + kk + khalf * 8];
                    v8h lo = *(const v8h*)pa;          // K kk+kh*8 .. +7
                    v8h hi = *(const v8h*)(pa + 16);   // K kk+16+kh*8 .. +7
                    aF[im] = __builtin_shufflevector(lo, hi,
                        0,1,2,3,4,5,6,7,8,9,10,11,12,13,14,15);
                }
                // B fragments: 32x16 f16, same K order as A: lanes 0-15 K kk..kk+7 and kk+16..kk+23, lanes 16-31 +8
                v16h bF[4];
                #pragma unroll
                for (int in = 0; in < 4; ++in) {
                    const _Float16* pb =
                        &Wt[(wave_n * 64 + in * 16 + lrow) * WPITCH + kk + khalf * 8];
                    v8h lo = *(const v8h*)pb;
                    v8h hi = *(const v8h*)(pb + 16);
                    bF[in] = __builtin_shufflevector(lo, hi,
                        0,1,2,3,4,5,6,7,8,9,10,11,12,13,14,15);
                }
                #pragma unroll
                for (int im = 0; im < 2; ++im)
                    #pragma unroll
                    for (int in = 0; in < 4; ++in)
                        acc[im][in] = __builtin_amdgcn_wmma_f32_16x16x32_f16(
                            false, aF[im], false, bF[in],
                            (short)0, acc[im][in], false, false);
            }
        }
    }

    // ---- Epilogue: C/D layout -> global f32 ----
    // VGPR r: lanes 0-15 hold M=r, lanes 16-31 hold M=8+r; N = lane%16
    const int lrow   = lane & 15;
    const int rowoff = (lane >> 4) * 8;
    #pragma unroll
    for (int im = 0; im < 2; ++im) {
        #pragma unroll
        for (int in = 0; in < 4; ++in) {
            const int col = n0 + wave_n * 64 + in * 16 + lrow;
            #pragma unroll
            for (int r = 0; r < 8; ++r) {
                const int row = m0 + wave_m * 32 + im * 16 + rowoff + r;
                out[(size_t)row * N + col] = acc[im][in][r];
            }
        }
    }
}

extern "C" void kernel_launch(void* const* d_in, const int* in_sizes, int n_in,
                              void* d_out, int out_size, void* d_ws, size_t ws_size,
                              hipStream_t stream) {
    (void)in_sizes; (void)n_in; (void)out_size; (void)d_ws; (void)ws_size;
    const float*  x  = (const float*)d_in[0];          // (4,2048,4096) fp32
    const int8_t* w  = (const int8_t*)d_in[1];         // (32,128,4096) int8 -> (K,N)
    const float*  sc = (const float*)d_in[2];          // (32,4096) fp32
    float* out = (float*)d_out;                        // (4,2048,4096) fp32

    dim3 grid(N / BN, M_TOTAL / BM);                   // 32 x 64 workgroups
    wq_gemm_wmma<<<grid, dim3(256), 0, stream>>>(x, w, sc, out);
}
